// AdvancedSpikingNeuralNetwork_4681514352743
// MI455X (gfx1250) — matrix operand order repaired and checked by reading, not compiled, not run on hardware
//
#include <hip/hip_runtime.h>

// ---------------------------------------------------------------------------
// LIF spiking net, 3 layers:  proj = act @ W^T + b  ->  temporal scan over T.
// GEMMs use CDNA5 v_wmma_f32_16x16x32_f16 (f16 in, f32 accumulate).
// Spikes scaled by (1-inhibition)=0.5 are exact in f16, so inter-layer
// activations are stored as f16 with no representational loss.
// ---------------------------------------------------------------------------

typedef __attribute__((ext_vector_type(16))) _Float16 v16h;
typedef __attribute__((ext_vector_type(8)))  _Float16 v8h;
typedef __attribute__((ext_vector_type(8)))  float    v8f;

static constexpr int   BATCH = 64;
static constexpr int   T     = 1024;
static constexpr int   INDIM = 256;
static constexpr int   H     = 512;
static constexpr int   M     = BATCH * T;     // 65536 rows in every GEMM
static constexpr float ALPHA = 0.95122942450071400909f;  // exp(-1/20)

// ---------------------------------------------------------------------------
// Utility kernels
// ---------------------------------------------------------------------------
__global__ void f32_to_f16_kernel(const float* __restrict__ in,
                                  _Float16* __restrict__ out, int n) {
  int i = blockIdx.x * blockDim.x + threadIdx.x;
  int stride = gridDim.x * blockDim.x;
  for (; i < n; i += stride) out[i] = (_Float16)in[i];
}

__global__ void zero_f32_kernel(float* __restrict__ p, int n) {
  int i = blockIdx.x * blockDim.x + threadIdx.x;
  if (i < n) p[i] = 0.0f;
}

// ---------------------------------------------------------------------------
// WMMA GEMM:  C[M,N] = A[M,K] (f16 row-major) * W[N,K]^T (f16 row-major) + bias
// Per wave: 32(M) x 64(N) output tile -> 8 accumulators, 8 WMMAs per K-step.
// Block = 256 threads = 8 waves covering 8 consecutive 32-row M groups at the
// same 64-col N group (weight fragments hit L0/L2).
//
// Fragment layouts used here (wave32; A/B use contiguous k, same order in both):
//   A 16x32 f16 : lane l(0..15): M=l, K=0..15 in v0..7
//                 lane l+16   : M=l, K=16..31
//   B 32x16 f16 : lane l(0..15): N=l, K=0..15 in v0..7 ; lane l+16: K=16..31
//     (memory is W[n][k] row-major == B^T, so each lane loads 16 contiguous k)
//   C 16x16 f32 : vgpr j, lane l(0..15): (M=j,  N=l) ; lane l+16: (M=j+8, N=l-16)
// ---------------------------------------------------------------------------
__device__ __forceinline__ v16h load_a_frag(const _Float16* __restrict__ arow,
                                            int k0, int hi) {
  v8h a0 = *reinterpret_cast<const v8h*>(arow + k0 + hi * 16);
  v8h a1 = *reinterpret_cast<const v8h*>(arow + k0 + hi * 16 + 8);
  v16h a;
#pragma unroll
  for (int i = 0; i < 8; ++i) { a[i] = a0[i]; a[8 + i] = a1[i]; }
  return a;
}

__device__ __forceinline__ v16h load_b_frag(const _Float16* __restrict__ wrow) {
  v8h b0 = *reinterpret_cast<const v8h*>(wrow);
  v8h b1 = *reinterpret_cast<const v8h*>(wrow + 8);
  v16h b;
#pragma unroll
  for (int i = 0; i < 8; ++i) { b[i] = b0[i]; b[8 + i] = b1[i]; }
  return b;
}

__global__ void __launch_bounds__(256)
gemm_f16_wmma_kernel(const _Float16* __restrict__ A,
                     const _Float16* __restrict__ W,
                     const float* __restrict__ bias,
                     float* __restrict__ C,
                     int K, int N) {
  const int lane = threadIdx.x & 31;
  const int wave = threadIdx.x >> 5;
  const int l    = lane & 15;
  const int hi   = lane >> 4;

  const int mBase = (blockIdx.x * 8 + wave) * 32;   // 32 rows per wave
  const int nBase = blockIdx.y * 64;                // 64 cols per wave

  const _Float16* arow0 = A + (size_t)(mBase + l) * K;
  const _Float16* arow1 = A + (size_t)(mBase + 16 + l) * K;

  v8f acc[2][4] = {};

  for (int k0 = 0; k0 < K; k0 += 32) {
    v16h a0 = load_a_frag(arow0, k0, hi);
    v16h a1 = load_a_frag(arow1, k0, hi);
#pragma unroll
    for (int s = 0; s < 4; ++s) {
      const int ncol = nBase + s * 16 + l;
      const _Float16* wrow = W + (size_t)ncol * K + k0 + hi * 16;
      v16h bf = load_b_frag(wrow);
      acc[0][s] = __builtin_amdgcn_wmma_f32_16x16x32_f16(
          false, a0, false, bf, (short)0, acc[0][s], false, false);
      acc[1][s] = __builtin_amdgcn_wmma_f32_16x16x32_f16(
          false, a1, false, bf, (short)0, acc[1][s], false, false);
    }
  }

#pragma unroll
  for (int s = 0; s < 4; ++s) {
    const int col = nBase + s * 16 + l;
    const float bv = bias[col];
#pragma unroll
    for (int hm = 0; hm < 2; ++hm) {
#pragma unroll
      for (int j = 0; j < 8; ++j) {
        const int r = mBase + hm * 16 + hi * 8 + j;
        C[(size_t)r * N + col] = acc[hm][s][j] + bv;
      }
    }
  }
}

// ---------------------------------------------------------------------------
// LIF temporal scan: one thread per (b,h) channel, sequential over T.
// m = ALPHA*m + proj; spike = (m>=thr); m *= (1-spike); out = spike*(1-inh).
// Membrane carried across layers via `mem` buffer (as in reference).
// LAST=false -> write f16 scaled spikes (next layer's GEMM input).
// LAST=true  -> write f32 scaled spikes to d_out.
// ---------------------------------------------------------------------------
template <bool LAST>
__global__ void lif_scan_kernel(const float* __restrict__ proj,
                                const float* __restrict__ thr,
                                const float* __restrict__ inh,
                                float* __restrict__ mem,
                                _Float16* __restrict__ actOut,
                                float* __restrict__ fOut) {
  const int idx = blockIdx.x * blockDim.x + threadIdx.x;   // b*H + h
  if (idx >= BATCH * H) return;
  const int h = idx & (H - 1);
  const int b = idx >> 9;          // H == 512

  float m     = mem[idx];
  const float th    = thr[h];
  const float scale = 1.0f - inh[h];
  size_t base = (size_t)b * T * H + h;

  for (int t = 0; t < T; ++t) {
    const size_t off = base + (size_t)t * H;
    float p = proj[off];
    m = ALPHA * m + p;
    float spike = (m >= th) ? 1.0f : 0.0f;
    m *= (1.0f - spike);
    float o = spike * scale;
    if (LAST) fOut[off] = o;
    else      actOut[off] = (_Float16)o;
  }
  mem[idx] = m;
}

// ---------------------------------------------------------------------------
// Launch
// ---------------------------------------------------------------------------
extern "C" void kernel_launch(void* const* d_in, const int* in_sizes, int n_in,
                              void* d_out, int out_size, void* d_ws, size_t ws_size,
                              hipStream_t stream) {
  (void)in_sizes; (void)n_in; (void)out_size; (void)ws_size;

  const float* x    = (const float*)d_in[0];
  const float* W0   = (const float*)d_in[1];
  const float* b0   = (const float*)d_in[2];
  const float* W1   = (const float*)d_in[3];
  const float* b1   = (const float*)d_in[4];
  const float* W2   = (const float*)d_in[5];
  const float* b2   = (const float*)d_in[6];
  const float* thr0 = (const float*)d_in[7];
  const float* thr1 = (const float*)d_in[8];
  const float* thr2 = (const float*)d_in[9];
  const float* inh  = (const float*)d_in[10];
  float* out = (float*)d_out;

  // Workspace layout (bytes, all sizes 256-aligned). Total ~290 MB.
  char* ws = (char*)d_ws;
  const size_t szXh   = (size_t)M * INDIM * sizeof(_Float16);  // 32 MB
  const size_t szAct  = (size_t)M * H * sizeof(_Float16);      // 64 MB
  const size_t szProj = (size_t)M * H * sizeof(float);         // 128 MB
  const size_t szW0   = (size_t)H * INDIM * sizeof(_Float16);
  const size_t szW12  = (size_t)H * H * sizeof(_Float16);
  const size_t szMem  = (size_t)BATCH * H * sizeof(float);

  _Float16* xh   = (_Float16*)(ws);
  _Float16* actA = (_Float16*)(ws + szXh);
  _Float16* actB = (_Float16*)(ws + szXh + szAct);
  float*    proj = (float*)   (ws + szXh + 2 * szAct);
  _Float16* W0h  = (_Float16*)(ws + szXh + 2 * szAct + szProj);
  _Float16* W1h  = (_Float16*)(ws + szXh + 2 * szAct + szProj + szW0);
  _Float16* W2h  = (_Float16*)(ws + szXh + 2 * szAct + szProj + szW0 + szW12);
  float*    mem  = (float*)   (ws + szXh + 2 * szAct + szProj + szW0 + 2 * szW12);
  (void)szMem;

  // 0) zero membrane, convert inputs/weights to f16
  zero_f32_kernel<<<(BATCH * H + 255) / 256, 256, 0, stream>>>(mem, BATCH * H);
  f32_to_f16_kernel<<<4096, 256, 0, stream>>>(x,  xh,  M * INDIM);
  f32_to_f16_kernel<<<512,  256, 0, stream>>>(W0, W0h, H * INDIM);
  f32_to_f16_kernel<<<1024, 256, 0, stream>>>(W1, W1h, H * H);
  f32_to_f16_kernel<<<1024, 256, 0, stream>>>(W2, W2h, H * H);

  const dim3 gemmGrid(M / (32 * 8), H / 64);   // (256, 8)
  const dim3 gemmBlock(256);
  const int  scanBlocks = (BATCH * H + 255) / 256;  // 128

  // Layer 0
  gemm_f16_wmma_kernel<<<gemmGrid, gemmBlock, 0, stream>>>(xh, W0h, b0, proj, INDIM, H);
  lif_scan_kernel<false><<<scanBlocks, 256, 0, stream>>>(proj, thr0, inh, mem, actA, nullptr);

  // Layer 1
  gemm_f16_wmma_kernel<<<gemmGrid, gemmBlock, 0, stream>>>(actA, W1h, b1, proj, H, H);
  lif_scan_kernel<false><<<scanBlocks, 256, 0, stream>>>(proj, thr1, inh, mem, actB, nullptr);

  // Layer 2
  gemm_f16_wmma_kernel<<<gemmGrid, gemmBlock, 0, stream>>>(actB, W2h, b2, proj, H, H);
  lif_scan_kernel<true><<<scanBlocks, 256, 0, stream>>>(proj, thr2, inh, mem, nullptr, out);
}
